// CausalAttention_83494164234316
// MI455X (gfx1250) — matrix operand order repaired and checked by reading, not compiled, not run on hardware
//
#include <hip/hip_runtime.h>

typedef _Float16 v16h __attribute__((ext_vector_type(16)));
typedef _Float16 v8h  __attribute__((ext_vector_type(8)));
typedef float    v8f  __attribute__((ext_vector_type(8)));
typedef unsigned int v8u __attribute__((ext_vector_type(8)));

#define N_SEQ 4096
#define C_IN  256
#define CH_T  256
#define DH_T  32

static __device__ __forceinline__ v16h mk16(v8h lo, v8h hi) {
    v16h r;
#pragma unroll
    for (int i = 0; i < 8; ++i) { r[i] = lo[i]; r[i + 8] = hi[i]; }
    return r;
}

static __device__ __forceinline__ unsigned pack2(float a, float b) {
    unsigned short ua = __builtin_bit_cast(unsigned short, (_Float16)a);
    unsigned short ub = __builtin_bit_cast(unsigned short, (_Float16)b);
    return (unsigned)ua | ((unsigned)ub << 16);
}

// ---------------- weight-norm: w[r,:] = g[r] * v[r,:] / ||v[r,:]|| ----------
__global__ __launch_bounds__(256) void wnorm_kernel(
    const float* __restrict__ vq, const float* __restrict__ gq,
    const float* __restrict__ vk, const float* __restrict__ gk,
    const float* __restrict__ vv, const float* __restrict__ gv,
    _Float16* __restrict__ wq, _Float16* __restrict__ wk, _Float16* __restrict__ wv) {
    int w    = blockIdx.x * (blockDim.x >> 5) + (threadIdx.x >> 5);
    int lane = threadIdx.x & 31;
    int mat  = w >> 8;
    int row  = w & 255;
    const float* v = (mat == 0) ? vq : (mat == 1 ? vk : vv);
    const float* g = (mat == 0) ? gq : (mat == 1 ? gk : gv);
    _Float16*    o = (mat == 0) ? wq : (mat == 1 ? wk : wv);

    const float* vr = v + row * C_IN + lane * 8;
    float x[8];
    float ss = 0.f;
#pragma unroll
    for (int i = 0; i < 8; ++i) { x[i] = vr[i]; ss += x[i] * x[i]; }
#pragma unroll
    for (int off = 16; off >= 1; off >>= 1) ss += __shfl_xor(ss, off, 32);
    float s = g[row] / sqrtf(ss);
    v8h out;
#pragma unroll
    for (int i = 0; i < 8; ++i) out[i] = (_Float16)(x[i] * s);
    *(v8h*)(o + row * C_IN + lane * 8) = out;
}

// ------------- [C][N] f32 -> [N][C] f16 transpose+convert (LDS tiled) -------
__global__ __launch_bounds__(256) void t16_kernel(
    const float* __restrict__ query, const float* __restrict__ keyf,
    _Float16* __restrict__ qT, _Float16* __restrict__ kT) {
    __shared__ float t[32][33];
    const float* in  = blockIdx.z ? keyf : query;
    _Float16*    out = blockIdx.z ? kT : qT;
    int n0 = blockIdx.x * 32, c0 = blockIdx.y * 32;
    int tid = threadIdx.x;
    int nl = tid & 31, cl = tid >> 5;
#pragma unroll
    for (int j = 0; j < 4; ++j)
        t[cl + 8 * j][nl] = in[(size_t)(c0 + cl + 8 * j) * N_SEQ + n0 + nl];
    __syncthreads();
    int cl2 = tid & 31, nl2 = tid >> 5;
#pragma unroll
    for (int j = 0; j < 4; ++j)
        out[(size_t)(n0 + nl2 + 8 * j) * C_IN + c0 + cl2] = (_Float16)t[cl2][nl2 + 8 * j];
}

// ------------- projections: Q/K -> [N][CH] f16, V -> [CH][N] f16 ------------
__global__ __launch_bounds__(256) void proj_kernel(
    const _Float16* __restrict__ qT, const _Float16* __restrict__ kT,
    const _Float16* __restrict__ wq, const _Float16* __restrict__ wk,
    const _Float16* __restrict__ wv,
    const float* __restrict__ bq, const float* __restrict__ bk,
    const float* __restrict__ bv,
    _Float16* __restrict__ Qh, _Float16* __restrict__ Kh, _Float16* __restrict__ Vt) {
    int mat  = blockIdx.y;
    int w    = blockIdx.x * 8 + (threadIdx.x >> 5);
    int lane = threadIdx.x & 31;
    int n0   = (w & 255) * 16;
    int ch0  = (w >> 8) * 16;

    const _Float16* X  = (mat == 0) ? qT : kT;
    const _Float16* Wm = (mat == 0) ? wq : (mat == 1 ? wk : wv);
    const float*    bb = (mat == 0) ? bq : (mat == 1 ? bk : bv);

    int idx   = lane & 15;
    int hi16  = lane >> 4;
    int abase = hi16 << 3;   // A-layout per-lane K base: 0 / 8
    int bbase = hi16 << 4;   // contiguous-order K base: 0 / 16 (unused; B also loads at abase)

    v8f acc = {};
#pragma unroll
    for (int c0 = 0; c0 < C_IN; c0 += 32) {
        const _Float16* pa = X + (size_t)(n0 + idx) * C_IN + c0 + abase;
        v16h a = mk16(*(const v8h*)pa, *(const v8h*)(pa + 16));
        const _Float16* pb = Wm + (size_t)(ch0 + idx) * C_IN + c0 + abase;
        v16h b = mk16(*(const v8h*)pb, *(const v8h*)(pb + 16));
        acc = __builtin_amdgcn_wmma_f32_16x16x32_f16(false, a, false, b,
                                                     (short)0, acc, false, false);
    }
    float bs  = bb[ch0 + idx];
    int moff  = abase;  // C-layout row base per half-wave
    if (mat == 2) {
        v8h ov;
#pragma unroll
        for (int r = 0; r < 8; ++r) ov[r] = (_Float16)(acc[r] + bs);
        *(v8h*)(Vt + (size_t)(ch0 + idx) * N_SEQ + n0 + moff) = ov;
    } else {
        float sc = (mat == 0) ? 0.17677669529663687f : 1.0f;  // 1/sqrt(32) folded into Q
        _Float16* O = (mat == 0) ? Qh : Kh;
#pragma unroll
        for (int r = 0; r < 8; ++r)
            O[(size_t)(n0 + moff + r) * CH_T + ch0 + idx] = (_Float16)((acc[r] + bs) * sc);
    }
}

// ---- flash attention, S computed transposed: S'[kj][qi] = (K Q^T) ----------
__global__ __launch_bounds__(256) void attn_kernel(
    const _Float16* __restrict__ Qh, const _Float16* __restrict__ Kh,
    const _Float16* __restrict__ Vt, float* __restrict__ out) {
    int w    = blockIdx.x * 8 + (threadIdx.x >> 5);
    int lane = threadIdx.x & 31;
    int head = w >> 8;
    int q0   = (w & 255) * 16;
    int idx  = lane & 15;
    int hi16 = lane >> 4;
    int abase = hi16 << 3;
    int bbase = hi16 << 4;
    int qi = q0 + idx;

    // Q as B operand: B[d][qi] = Q[q0+qi][head*32+d]
    const _Float16* pq = Qh + (size_t)qi * CH_T + head * DH_T + abase;
    v16h qb = mk16(*(const v8h*)pq, *(const v8h*)(pq + 16));

    v8f acc0 = {}, acc1 = {};
    float m = -INFINITY, l = 0.f;

    for (int k0 = 0; k0 < q0 + 15; k0 += 32) {
        // K tiles as A operand: A[kj][d]
        const _Float16* pk = Kh + (size_t)(k0 + idx) * CH_T + head * DH_T + abase;
        v16h ka = mk16(*(const v8h*)pk, *(const v8h*)(pk + 16));
        const _Float16* pk2 = pk + (size_t)16 * CH_T;
        v16h kb = mk16(*(const v8h*)pk2, *(const v8h*)(pk2 + 16));

        v8f z = {};
        v8f sa = __builtin_amdgcn_wmma_f32_16x16x32_f16(false, ka, false, qb,
                                                        (short)0, z, false, false);
        v8f sb = __builtin_amdgcn_wmma_f32_16x16x32_f16(false, kb, false, qb,
                                                        (short)0, z, false, false);
        // strict causal mask: valid iff kj < qi
        if (k0 + 31 >= q0) {
            int kbase = k0 + abase;
#pragma unroll
            for (int r = 0; r < 8; ++r) {
                if (kbase + r >= qi)      sa[r] = -INFINITY;
                if (kbase + 16 + r >= qi) sb[r] = -INFINITY;
            }
        }
        // online softmax (per lane: fixed qi; partner lane L^16 holds other half)
        float tm = -INFINITY;
#pragma unroll
        for (int r = 0; r < 8; ++r) { tm = fmaxf(tm, sa[r]); tm = fmaxf(tm, sb[r]); }
        tm = fmaxf(tm, __shfl_xor(tm, 16, 32));
        float mn = fmaxf(m, tm);
        float ms = (mn == -INFINITY) ? 0.f : mn;
        float sc = __expf(m - ms);
        float pa[8], pb[8], ts = 0.f;
#pragma unroll
        for (int r = 0; r < 8; ++r) {
            pa[r] = __expf(sa[r] - ms);
            pb[r] = __expf(sb[r] - ms);
            ts += pa[r] + pb[r];
        }
        ts += __shfl_xor(ts, 16, 32);
        l = l * sc + ts;
        m = mn;
#pragma unroll
        for (int r = 0; r < 8; ++r) { acc0[r] *= sc; acc1[r] *= sc; }

        // pack P into B operand layout (table order: local pa K 8h..8h+7 + local pb K 16+8h..16+8h+7)
        unsigned pka[4], pkb[4];
#pragma unroll
        for (int j = 0; j < 4; ++j) {
            pka[j] = pack2(pa[2 * j], pa[2 * j + 1]);
            pkb[j] = pack2(pb[2 * j], pb[2 * j + 1]);
        }
        v8u bw;
#pragma unroll
        for (int j = 0; j < 4; ++j) {
            unsigned oa = (unsigned)__shfl_xor((int)pka[j], 16, 32);
            unsigned ob = (unsigned)__shfl_xor((int)pkb[j], 16, 32);
            bw[j]     = pka[j];
            bw[j + 4] = pkb[j];
        }
        v16h pf = __builtin_bit_cast(v16h, bw);

        // V^T tiles as A operand: A[d][kidx] = Vt[head*32+d][k0+kidx]
        const _Float16* pv = Vt + (size_t)(head * DH_T + idx) * N_SEQ + k0 + abase;
        v16h va = mk16(*(const v8h*)pv, *(const v8h*)(pv + 16));
        const _Float16* pv2 = pv + (size_t)16 * N_SEQ;
        v16h vb = mk16(*(const v8h*)pv2, *(const v8h*)(pv2 + 16));

        acc0 = __builtin_amdgcn_wmma_f32_16x16x32_f16(false, va, false, pf,
                                                      (short)0, acc0, false, false);
        acc1 = __builtin_amdgcn_wmma_f32_16x16x32_f16(false, vb, false, pf,
                                                      (short)0, acc1, false, false);
    }
    float inv = (l > 0.f) ? 1.f / l : 0.f;  // row 0 fully masked -> 0 (start_mask)
#pragma unroll
    for (int r = 0; r < 8; ++r) {
        out[(size_t)(head * DH_T + abase + r) * N_SEQ + qi]      = acc0[r] * inv;
        out[(size_t)(head * DH_T + 16 + abase + r) * N_SEQ + qi] = acc1[r] * inv;
    }
}

extern "C" void kernel_launch(void* const* d_in, const int* in_sizes, int n_in,
                              void* d_out, int out_size, void* d_ws, size_t ws_size,
                              hipStream_t stream) {
    (void)in_sizes; (void)n_in; (void)out_size; (void)ws_size;
    const float* query = (const float*)d_in[0];
    const float* keyf  = (const float*)d_in[1];
    const float* vq = (const float*)d_in[2];
    const float* gq = (const float*)d_in[3];
    const float* bq = (const float*)d_in[4];
    const float* vk = (const float*)d_in[5];
    const float* gk = (const float*)d_in[6];
    const float* bk = (const float*)d_in[7];
    const float* vv = (const float*)d_in[8];
    const float* gv = (const float*)d_in[9];
    const float* bv = (const float*)d_in[10];

    char* ws = (char*)d_ws;
    const size_t WSZ = (size_t)256 * 256 * sizeof(_Float16);      // weights
    const size_t TSZ = (size_t)N_SEQ * 256 * sizeof(_Float16);    // activations
    _Float16* wq = (_Float16*)ws;            ws += WSZ;
    _Float16* wk = (_Float16*)ws;            ws += WSZ;
    _Float16* wv = (_Float16*)ws;            ws += WSZ;
    _Float16* qT = (_Float16*)ws;            ws += TSZ;
    _Float16* kT = (_Float16*)ws;            ws += TSZ;
    _Float16* Qh = (_Float16*)ws;            ws += TSZ;
    _Float16* Kh = (_Float16*)ws;            ws += TSZ;
    _Float16* Vt = (_Float16*)ws;            ws += TSZ;

    wnorm_kernel<<<96, 256, 0, stream>>>(vq, gq, vk, gk, vv, gv, wq, wk, wv);
    t16_kernel<<<dim3(N_SEQ / 32, 256 / 32, 2), 256, 0, stream>>>(query, keyf, qT, kT);
    proj_kernel<<<dim3(512, 3), 256, 0, stream>>>(qT, kT, wq, wk, wv, bq, bk, bv,
                                                  Qh, Kh, Vt);
    attn_kernel<<<256, 256, 0, stream>>>(Qh, Kh, Vt, (float*)d_out);
}
